// DgcnnSegment_43619687858603
// MI455X (gfx1250) — matrix operand order repaired and checked by reading, not compiled, not run on hardware
//
#include <hip/hip_runtime.h>

typedef __attribute__((ext_vector_type(16))) __bf16 v16bf;
typedef __attribute__((ext_vector_type(8)))  float  v8f;
typedef __attribute__((ext_vector_type(4)))  float  f4;
typedef __attribute__((ext_vector_type(4)))  unsigned int u4;
typedef __attribute__((ext_vector_type(8)))  int    i8v;
typedef __attribute__((ext_vector_type(4)))  int    i4v;

#define BQ 4
#define NQ 4096
#define CQ 128
#define KNN 10
#define BN_ROWS (BQ * NQ)   // 16384

// EdgeConv weights packed contiguously per block: W1t(128x128)|W2t(64x128)|W3t(64x64)
#define EC_W1_ELEMS  (128 * 128)                           // 16384
#define EC_W2_ELEMS  (64 * 128)                            // 8192
#define EC_W3_ELEMS  (64 * 64)                             // 4096
#define EC_W_ELEMS   (EC_W1_ELEMS + EC_W2_ELEMS + EC_W3_ELEMS)  // 28672
#define EC_STAGE     (16 * 128 + 16 * 64)                  // 3072 per wave
#define EC_LDS_BYTES ((EC_W_ELEMS + 8 * EC_STAGE) * 2)     // 106496

// ---------------------------------------------------------------------------
// WMMA helpers (bf16 16x16x32, f32 accum) — layouts per CDNA5 ISA 7.12.2
// ---------------------------------------------------------------------------
__device__ __forceinline__ v8f wmma_bf16(v16bf a, v16bf b, v8f c) {
  return __builtin_amdgcn_wmma_f32_16x16x32_bf16(false, a, false, b, (short)0, c, false, false);
}

// 16 contiguous bf16 -> fragment register block (two b128 loads)
__device__ __forceinline__ v16bf load16bf(const __bf16* p) {
  union { u4 q[2]; v16bf v; } u;
  u.q[0] = *(const u4*)p;
  u.q[1] = *(const u4*)(p + 8);
  return u.v;
}

// A fragment 16x32 bf16 from row-major storage (lda elements, 16B-aligned rows).
// lane<16: row=lane,     K = 0..7  and 16..23
// lane>=16: row=lane-16, K = 8..15 and 24..31
__device__ __forceinline__ v16bf load_a_rm(const __bf16* A, int lda, int lane) {
  int row = lane & 15;
  int kg  = (lane >> 4) << 3;
  const __bf16* p = A + row * lda + kg;
  union { u4 q[2]; v16bf v; } u;
  u.q[0] = *(const u4*)p;         // K offsets +0..7
  u.q[1] = *(const u4*)(p + 16);  // K offsets +16..23
  return u.v;
}

// B fragment from TRANSPOSED weights Wt[N][K] (row-major, ldk elements):
// lane<16: col=lane, K=k0+0..7 and k0+16..23 ; lane>=16: col=lane-16, K=k0+8..15 and k0+24..31
__device__ __forceinline__ v16bf load_b_tr(const __bf16* Wt, int ldk, int col0,
                                           int k0, int lane) {
  int col = lane & 15;
  int kg  = k0 + ((lane >> 4) << 3);
  const __bf16* p = Wt + (size_t)(col0 + col) * ldk + kg;
  union { u4 q[2]; v16bf v; } u;
  u.q[0] = *(const u4*)p;
  u.q[1] = *(const u4*)(p + 16);
  return u.v;
}

// ---------------------------------------------------------------------------
// TDM: 1-D tensor_load_to_lds of `elems` bf16 from `gsrc` to LDS `ldst`.
// Descriptor layout per CDNA5 ISA.  Issue from one
// wave only (TDM ignores EXEC); caller must __syncthreads() afterwards.
// ---------------------------------------------------------------------------
__device__ __forceinline__ void tdm_load_1d(const __bf16* gsrc, __bf16* ldst,
                                            unsigned elems) {
  unsigned long long ga = (unsigned long long)(uintptr_t)gsrc;
  unsigned lds_addr = (unsigned)(uintptr_t)ldst;   // LDS aperture: addr[31:0]
  u4 g0;
  g0[0] = 1u;                                       // count=1 valid descriptor
  g0[1] = lds_addr;                                 // lds_addr[31:0]
  g0[2] = (unsigned)(ga & 0xffffffffu);             // global_addr[31:0]
  g0[3] = (unsigned)((ga >> 32) & 0x01ffffffu)      // global_addr[56:32]
        | 0x80000000u;                              // type=2 ("image")
  i8v g1;
  g1[0] = 0x00010000;                               // data_size=1 (2B), mask=0
  g1[1] = (int)((elems & 0xffffu) << 16);           // tensor_dim0[15:0]
  g1[2] = (int)((elems >> 16) | (1u << 16));        // tensor_dim0[31:16], tensor_dim1=1
  g1[3] = (int)((elems & 0xffffu) << 16);           // tile_dim0 (1-D tile)
  g1[4] = 0;                                        // tile_dim1=0 (unused), tile_dim2=0
  g1[5] = (int)elems;                               // tensor_dim0_stride[31:0]
  g1[6] = 0;
  g1[7] = 0;
  i4v gz = {};                                      // groups 2/3 unused (<=2-D)
#if defined(__clang_major__) && (__clang_major__ >= 23)
  i8v gz8 = {};
  __builtin_amdgcn_tensor_load_to_lds(g0, g1, gz, gz, gz8, 0);
#else
  __builtin_amdgcn_tensor_load_to_lds(g0, g1, gz, gz, 0);
#endif
  __builtin_amdgcn_s_wait_tensorcnt(0);
}

// ---------------------------------------------------------------------------
// fp32 [K][N] -> bf16 [N][K] transpose-convert (weights into B-frag layout)
// grid: (ceil(N/256), K)
// ---------------------------------------------------------------------------
__global__ void cvt_tr_kernel(const float* __restrict__ src, __bf16* __restrict__ dst,
                              int K, int N) {
  int n = blockIdx.x * 256 + threadIdx.x;
  int k = blockIdx.y;
  if (n < N) dst[n * K + k] = (__bf16)src[k * N + n];
}

// ---------------------------------------------------------------------------
// BatchNorm statistics folded to per-channel scale/shift
// ---------------------------------------------------------------------------
__global__ void bn_stats_kernel(const float* __restrict__ feat,
                                const float* __restrict__ gamma,
                                const float* __restrict__ beta,
                                float* __restrict__ scale, float* __restrict__ shift) {
  __shared__ float s1[256], s2[256];
  int c = blockIdx.x, t = threadIdx.x;
  float s = 0.f, sq = 0.f;
  for (int r = t; r < BN_ROWS; r += 256) {
    float v = feat[r * CQ + c];
    s += v; sq += v * v;
  }
  s1[t] = s; s2[t] = sq;
  __syncthreads();
  for (int o = 128; o > 0; o >>= 1) {
    if (t < o) { s1[t] += s1[t + o]; s2[t] += s2[t + o]; }
    __syncthreads();
  }
  if (t == 0) {
    float m = s1[0] * (1.0f / BN_ROWS);
    float v = s2[0] * (1.0f / BN_ROWS) - m * m;
    float sc = rsqrtf(v + 1e-5f) * gamma[c];
    scale[c] = sc;
    shift[c] = beta[c] - m * sc;
  }
}

// ---------------------------------------------------------------------------
// Fused BN-normalize + pre-linear (128->64) + ReLU.  Wave per 16 rows.
// ---------------------------------------------------------------------------
__global__ void pre_kernel(const float* __restrict__ feat,
                           const float* __restrict__ scl, const float* __restrict__ sft,
                           const __bf16* __restrict__ Wt, const float* __restrict__ bias,
                           float* __restrict__ x0f, __bf16* __restrict__ x0b) {
  int lane = threadIdx.x & 31;
  int wave = threadIdx.x >> 5;
  int row0 = (blockIdx.x * 8 + wave) * 16;
  int col  = lane & 15;
  int row  = lane & 15;
  int kg   = (lane >> 4) << 3;
  const float* fr = feat + (size_t)(row0 + row) * CQ;

  v8f acc[4];
#pragma unroll
  for (int nt = 0; nt < 4; ++nt) {
    float bv = bias[nt * 16 + col];
#pragma unroll
    for (int r = 0; r < 8; ++r) acc[nt][r] = bv;
  }
  for (int k0 = 0; k0 < 128; k0 += 32) {
    v16bf a;
    int f0 = k0 + kg;
#pragma unroll
    for (int h = 0; h < 2; ++h) {
      int f = f0 + h * 16;
      f4 xa = *(const f4*)(fr + f),  xb = *(const f4*)(fr + f + 4);
      f4 sa = *(const f4*)(scl + f), sb = *(const f4*)(scl + f + 4);
      f4 ta = *(const f4*)(sft + f), tb = *(const f4*)(sft + f + 4);
#pragma unroll
      for (int i = 0; i < 4; ++i) {
        a[h * 8 + i]     = (__bf16)(xa[i] * sa[i] + ta[i]);
        a[h * 8 + 4 + i] = (__bf16)(xb[i] * sb[i] + tb[i]);
      }
    }
#pragma unroll
    for (int nt = 0; nt < 4; ++nt)
      acc[nt] = wmma_bf16(a, load_b_tr(Wt, 128, nt * 16, k0, lane), acc[nt]);
  }
  int r0 = (lane >> 4) << 3;
#pragma unroll
  for (int nt = 0; nt < 4; ++nt)
#pragma unroll
    for (int r = 0; r < 8; ++r) {
      float v = fmaxf(acc[nt][r], 0.f);
      int rr = row0 + r0 + r, cc = nt * 16 + col;
      x0f[(size_t)rr * 64 + cc] = v;
      x0b[(size_t)rr * 64 + cc] = (__bf16)v;
    }
}

// ---------------------------------------------------------------------------
// Squared norms per point (D=64)
// ---------------------------------------------------------------------------
__global__ void sqnorm_kernel(const float* __restrict__ x, float* __restrict__ sq) {
  int p = blockIdx.x * 256 + threadIdx.x;
  if (p >= BN_ROWS) return;
  float s = 0.f;
#pragma unroll
  for (int i = 0; i < 16; ++i) {
    f4 v = *(const f4*)(x + (size_t)p * 64 + i * 4);
    s += v[0] * v[0] + v[1] * v[1] + v[2] * v[2] + v[3] * v[3];
  }
  sq[p] = s;
}

// ---------------------------------------------------------------------------
// kNN: ONE WAVE per workgroup (barriers are S_NOP), 16 query rows per wave.
// 64 columns per iteration via 8 WMMAs; ||j||^2 folded into LDS tile at store
// time; ||i||^2 dropped (row-constant shift does not change top-k ranking).
// ---------------------------------------------------------------------------
__global__ void knn_kernel(const __bf16* __restrict__ xb, const float* __restrict__ sq,
                           int* __restrict__ knn) {
  __shared__ __align__(16) float tl[4][16][17];
  int lane  = threadIdx.x & 31;
  int b     = blockIdx.x >> 8;            // 256 blocks per batch
  int qrow0 = (blockIdx.x & 255) * 16;    // within batch
  const __bf16* X  = xb + (size_t)b * NQ * 64;
  const float*  SQ = sq + (size_t)b * NQ;

  v16bf a0 = load_a_rm(X + qrow0 * 64, 64, lane);
  v16bf a1 = load_a_rm(X + qrow0 * 64 + 32, 64, lane);

  float dv[KNN]; int iv[KNN];
#pragma unroll
  for (int i = 0; i < KNN; ++i) { dv[i] = 3.402823e38f; iv[i] = 0; }
  int r0 = (lane >> 4) << 3, c = lane & 15;

  for (int j0 = 0; j0 < NQ; j0 += 64) {
#pragma unroll
    for (int t = 0; t < 4; ++t) {
      int jt = j0 + t * 16;
      v8f acc = {};
      acc = wmma_bf16(a0, load_a_rm(X + (size_t)jt * 64, 64, lane), acc);        // K 0..31 by lane half
      acc = wmma_bf16(a1, load_a_rm(X + (size_t)jt * 64 + 32, 64, lane), acc);   // K 32..63
      float sqj = SQ[jt + c];
#pragma unroll
      for (int r = 0; r < 8; ++r) tl[t][r0 + r][c] = sqj - 2.f * acc[r];
    }
    __syncthreads();
    if (lane < 16) {
      // partial unroll: keeps insert-chain register indices static, tames I-cache
#pragma unroll 1
      for (int t = 0; t < 4; ++t)
#pragma unroll 4
        for (int c2 = 0; c2 < 16; ++c2) {
          float d = tl[t][lane][c2];
          if (d < dv[KNN - 1]) {
            float cd = d; int ci = j0 + t * 16 + c2;
#pragma unroll
            for (int p = 0; p < KNN; ++p) {
              if (cd < dv[p]) {
                float td = dv[p]; int ti = iv[p];
                dv[p] = cd; iv[p] = ci; cd = td; ci = ti;
              }
            }
          }
        }
    }
    __syncthreads();
  }
  if (lane < 16) {
    int q = b * NQ + qrow0 + lane;
#pragma unroll
    for (int i = 0; i < KNN; ++i) knn[q * KNN + i] = iv[i];
  }
}

// ---------------------------------------------------------------------------
// EdgeConv: 8 waves / block; packed weights DMA'd ONCE into LDS by the TDM
// and reused by 128 points per block (16 per wave).  Edges padded 10->16
// rows; MLP 128->128->64->64 via WMMA; inter-layer staging in wave-private LDS.
// ---------------------------------------------------------------------------
__global__ void edgeconv_kernel(const float* __restrict__ xin, const int* __restrict__ knn,
                                const __bf16* __restrict__ Wall,   // W1t|W2t|W3t packed
                                const float* __restrict__ b1,
                                const float* __restrict__ b2,
                                const float* __restrict__ b3,
                                float* __restrict__ outf, __bf16* __restrict__ outb,
                                int ostride, int ocol) {
  extern __shared__ __bf16 smem[];
  __bf16* w1 = smem;                          // [128][128]
  __bf16* w2 = w1 + EC_W1_ELEMS;              // [64][128]
  __bf16* w3 = w2 + EC_W2_ELEMS;              // [64][64]
  int lane = threadIdx.x & 31, wave = threadIdx.x >> 5;
  __bf16* h1 = w3 + EC_W3_ELEMS + wave * EC_STAGE;  // [16][128]
  __bf16* h2 = h1 + 16 * 128;                       // [16][64]

  // async weight staging via Tensor Data Mover (one DMA for all 57344 B)
  if (threadIdx.x == 0) tdm_load_1d(Wall, smem, EC_W_ELEMS);
  __syncthreads();

  int row = lane & 15, col = lane & 15;
  int kg  = (lane >> 4) << 3;
  int r0  = (lane >> 4) << 3;
  bool valid = row < KNN;

  // hoisted biases
  float bv1[8], bv2[4], bv3[4];
#pragma unroll
  for (int nt = 0; nt < 8; ++nt) bv1[nt] = b1[nt * 16 + col];
#pragma unroll
  for (int nt = 0; nt < 4; ++nt) bv2[nt] = b2[nt * 16 + col];
#pragma unroll
  for (int nt = 0; nt < 4; ++nt) bv3[nt] = b3[nt * 16 + col];

  int pbase = blockIdx.x * 128 + wave * 16;
  for (int pi = 0; pi < 16; ++pi) {
    int p = pbase + pi;
    int bbase = (p >> 12) << 12;     // batch base point
    const float* xi = xin + (size_t)p * 64;
    const float* xj = xi;
    if (valid) {
      int j = knn[p * KNN + row];
      xj = xin + (size_t)(bbase + j) * 64;
    }

    // ---- layer 1: [16x128] @ [128x128] + b1, ReLU
    v8f acc[8];
#pragma unroll
    for (int nt = 0; nt < 8; ++nt)
#pragma unroll
      for (int r = 0; r < 8; ++r) acc[nt][r] = bv1[nt];
    for (int k0 = 0; k0 < 128; k0 += 32) {
      v16bf a;
      int f0 = k0 + kg;
#pragma unroll
      for (int h = 0; h < 2; ++h) {
        int f = f0 + h * 16;           // multiple of 8: run stays in one region
        f4 va = {}, vb = {};
        if (valid) {
          if (f < 64) {
            va = *(const f4*)(xi + f);
            vb = *(const f4*)(xi + f + 4);
          } else {
            int g = f - 64;
            f4 ja = *(const f4*)(xj + g), jb = *(const f4*)(xj + g + 4);
            f4 ia = *(const f4*)(xi + g), ib = *(const f4*)(xi + g + 4);
            va = ja - ia; vb = jb - ib;
          }
        }
#pragma unroll
        for (int i = 0; i < 4; ++i) {
          a[h * 8 + i]     = (__bf16)va[i];
          a[h * 8 + 4 + i] = (__bf16)vb[i];
        }
      }
#pragma unroll
      for (int nt = 0; nt < 8; ++nt)
        acc[nt] = wmma_bf16(a, load_b_tr(w1, 128, nt * 16, k0, lane), acc[nt]);
    }
#pragma unroll
    for (int nt = 0; nt < 8; ++nt)
#pragma unroll
      for (int r = 0; r < 8; ++r)
        h1[(r0 + r) * 128 + nt * 16 + col] = (__bf16)fmaxf(acc[nt][r], 0.f);
    __syncthreads();   // uniform trip counts across waves

    // ---- layer 2: [16x128] @ [128x64] + b2, ReLU
    v8f acc2[4];
#pragma unroll
    for (int nt = 0; nt < 4; ++nt)
#pragma unroll
      for (int r = 0; r < 8; ++r) acc2[nt][r] = bv2[nt];
    for (int k0 = 0; k0 < 128; k0 += 32) {
      v16bf a = load_a_rm(&h1[k0], 128, lane);
#pragma unroll
      for (int nt = 0; nt < 4; ++nt)
        acc2[nt] = wmma_bf16(a, load_b_tr(w2, 128, nt * 16, k0, lane), acc2[nt]);
    }
#pragma unroll
    for (int nt = 0; nt < 4; ++nt)
#pragma unroll
      for (int r = 0; r < 8; ++r)
        h2[(r0 + r) * 64 + nt * 16 + col] = (__bf16)fmaxf(acc2[nt][r], 0.f);
    __syncthreads();

    // ---- layer 3: [16x64] @ [64x64] + b3, ReLU, max over rows 0..9
    v8f acc3[4];
#pragma unroll
    for (int nt = 0; nt < 4; ++nt)
#pragma unroll
      for (int r = 0; r < 8; ++r) acc3[nt][r] = bv3[nt];
    for (int k0 = 0; k0 < 64; k0 += 32) {
      v16bf a = load_a_rm(&h2[k0], 64, lane);
#pragma unroll
      for (int nt = 0; nt < 4; ++nt)
        acc3[nt] = wmma_bf16(a, load_b_tr(w3, 64, nt * 16, k0, lane), acc3[nt]);
    }
#pragma unroll
    for (int nt = 0; nt < 4; ++nt) {
      float m = 0.f;                  // ReLU outputs are >= 0
      if (lane < 16) {                // rows 0..7
#pragma unroll
        for (int r = 0; r < 8; ++r) m = fmaxf(m, acc3[nt][r]);
      } else {                        // rows 8,9 (10..15 are padding)
        m = fmaxf(acc3[nt][0], acc3[nt][1]);
        m = fmaxf(m, 0.f);
      }
      float o = __shfl_xor(m, 16, 32);
      if (lane < 16) {
        float v = fmaxf(m, o);
        int cc = nt * 16 + lane;
        if (outf) outf[(size_t)p * 64 + cc] = v;
        outb[(size_t)p * ostride + ocol + cc] = (__bf16)v;
      }
    }
  }
}

// ---------------------------------------------------------------------------
// Final MLP layer: [16384x192] @ [192x192] + b, ReLU.  Wave per 16 rows.
// ---------------------------------------------------------------------------
__global__ void mlp2_kernel(const __bf16* __restrict__ A, const __bf16* __restrict__ Wt,
                            const float* __restrict__ bias,
                            float* __restrict__ outf, __bf16* __restrict__ outb) {
  int lane = threadIdx.x & 31, wave = threadIdx.x >> 5;
  int row0 = (blockIdx.x * 8 + wave) * 16;
  int col  = lane & 15;
  v8f acc[12];
#pragma unroll
  for (int nt = 0; nt < 12; ++nt) {
    float bv = bias[nt * 16 + col];
#pragma unroll
    for (int r = 0; r < 8; ++r) acc[nt][r] = bv;
  }
  for (int k0 = 0; k0 < 192; k0 += 32) {
    v16bf a = load_a_rm(A + (size_t)row0 * 192 + k0, 192, lane);
#pragma unroll
    for (int nt = 0; nt < 12; ++nt)
      acc[nt] = wmma_bf16(a, load_b_tr(Wt, 192, nt * 16, k0, lane), acc[nt]);
  }
  int r0 = (lane >> 4) << 3;
#pragma unroll
  for (int nt = 0; nt < 12; ++nt)
#pragma unroll
    for (int r = 0; r < 8; ++r) {
      float v = fmaxf(acc[nt][r], 0.f);
      int rr = row0 + r0 + r, cc = nt * 16 + col;
      if (outf) outf[(size_t)rr * 192 + cc] = v;
      else      outb[(size_t)rr * 192 + cc] = (__bf16)v;
    }
}

// ---------------------------------------------------------------------------
// Host orchestration
// ---------------------------------------------------------------------------
extern "C" void kernel_launch(void* const* d_in, const int* in_sizes, int n_in,
                              void* d_out, int out_size, void* d_ws, size_t ws_size,
                              hipStream_t stream) {
  (void)in_sizes; (void)n_in; (void)out_size; (void)ws_size;
  const float* feat   = (const float*)d_in[0];
  const float* gamma  = (const float*)d_in[1];
  const float* beta   = (const float*)d_in[2];
  const float* pre_W  = (const float*)d_in[3];
  const float* pre_b  = (const float*)d_in[4];
  const float* ecW1   = (const float*)d_in[5];
  const float* ecb1   = (const float*)d_in[6];
  const float* ecW2   = (const float*)d_in[7];
  const float* ecb2   = (const float*)d_in[8];
  const float* ecW3   = (const float*)d_in[9];
  const float* ecb3   = (const float*)d_in[10];
  const float* m2W1   = (const float*)d_in[11];
  const float* m2b1   = (const float*)d_in[12];
  const float* m2W2   = (const float*)d_in[13];
  const float* m2b2   = (const float*)d_in[14];

  char* w = (char*)d_ws;
  size_t off = 0;
  auto alloc = [&](size_t bytes) { size_t o = off; off = (off + bytes + 255) & ~(size_t)255; return o; };

  float*  scl   = (float*) (w + alloc(CQ * 4));
  float*  sft   = (float*) (w + alloc(CQ * 4));
  __bf16* wpreT = (__bf16*)(w + alloc(128 * 64 * 2));
  __bf16* wecT  = (__bf16*)(w + alloc(4 * EC_W_ELEMS * 2));   // packed W1t|W2t|W3t per block
  __bf16* wm1T  = (__bf16*)(w + alloc(192 * 192 * 2));
  __bf16* wm2T  = (__bf16*)(w + alloc(192 * 192 * 2));
  float*  x0f   = (float*) (w + alloc((size_t)BN_ROWS * 64 * 4));
  __bf16* x0b   = (__bf16*)(w + alloc((size_t)BN_ROWS * 64 * 2));
  float*  x1f   = (float*) (w + alloc((size_t)BN_ROWS * 64 * 4));
  __bf16* x1b   = (__bf16*)(w + alloc((size_t)BN_ROWS * 64 * 2));
  float*  sq0   = (float*) (w + alloc((size_t)BN_ROWS * 4));
  float*  sq1   = (float*) (w + alloc((size_t)BN_ROWS * 4));
  int*    idx0  = (int*)   (w + alloc((size_t)BN_ROWS * KNN * 4));
  int*    idx1  = (int*)   (w + alloc((size_t)BN_ROWS * KNN * 4));
  __bf16* fcat  = (__bf16*)(w + alloc((size_t)BN_ROWS * 192 * 2));
  __bf16* hmid  = (__bf16*)(w + alloc((size_t)BN_ROWS * 192 * 2));

  auto cvtT = [&](const float* s, __bf16* d, int K, int N) {
    dim3 g((N + 255) / 256, K);
    cvt_tr_kernel<<<g, 256, 0, stream>>>(s, d, K, N);
  };
  cvtT(pre_W, wpreT, 128, 64);
  for (int blk = 0; blk < 4; ++blk) {
    __bf16* base = wecT + (size_t)blk * EC_W_ELEMS;
    cvtT(ecW1 + blk * EC_W1_ELEMS, base,                             128, 128);
    cvtT(ecW2 + blk * EC_W2_ELEMS, base + EC_W1_ELEMS,               128, 64);
    cvtT(ecW3 + blk * EC_W3_ELEMS, base + EC_W1_ELEMS + EC_W2_ELEMS, 64,  64);
  }
  cvtT(m2W1, wm1T, 192, 192);
  cvtT(m2W2, wm2T, 192, 192);

  bn_stats_kernel<<<CQ, 256, 0, stream>>>(feat, gamma, beta, scl, sft);
  pre_kernel<<<128, 256, 0, stream>>>(feat, scl, sft, wpreT, pre_b, x0f, x0b);

  sqnorm_kernel<<<BN_ROWS / 256, 256, 0, stream>>>(x0f, sq0);
  knn_kernel<<<BQ * 256, 32, 0, stream>>>(x0b, sq0, idx0);

  // EdgeConv block 0 -> x1   (128 points per block, weights TDM->LDS resident)
  edgeconv_kernel<<<BN_ROWS / 128, 256, EC_LDS_BYTES, stream>>>(
      x0f, idx0, wecT, ecb1, ecb2, ecb3, x1f, x1b, 64, 0);

  sqnorm_kernel<<<BN_ROWS / 256, 256, 0, stream>>>(x1f, sq1);
  knn_kernel<<<BQ * 256, 32, 0, stream>>>(x1b, sq1, idx1);

  // Blocks 1..3 share x1's kNN graph; write bf16 into concat buffer columns
  for (int blk = 1; blk < 4; ++blk) {
    edgeconv_kernel<<<BN_ROWS / 128, 256, EC_LDS_BYTES, stream>>>(
        x1f, idx1,
        wecT + (size_t)blk * EC_W_ELEMS,
        ecb1 + blk * 128, ecb2 + blk * 64, ecb3 + blk * 64,
        nullptr, fcat, 192, (blk - 1) * 64);
  }

  mlp2_kernel<<<128, 256, 0, stream>>>(fcat, wm1T, m2b1, nullptr, hmid);
  mlp2_kernel<<<128, 256, 0, stream>>>(hmid, wm2T, m2b2, (float*)d_out, nullptr);
}
